// DotProductAttention_17377437680477
// MI455X (gfx1250) — matrix operand order repaired and checked by reading, not compiled, not run on hardware
//
#include <hip/hip_runtime.h>

typedef _Float16 v16h __attribute__((ext_vector_type(16)));
typedef float    v8f  __attribute__((ext_vector_type(8)));

#define NB    16
#define NN    2048
#define DD    64
#define SROW  2064          // padded row stride (words) to spread LDS banks
#define KEEP  1536          // top-k kept per row (N * 3/4)

__global__ __launch_bounds__(512, 1)
void attn_topk_kernel(const float* __restrict__ qg,
                      const float* __restrict__ kg,
                      const float* __restrict__ vg,
                      const int*   __restrict__ knnp,
                      float*       __restrict__ outg)
{
    __shared__ float    sc[16 * SROW];        // 16 x 2048 score strip (padded)
    __shared__ unsigned hist[16 * 256];       // per-wave radix histograms
    __shared__ unsigned thr_s[16];
    __shared__ float    rowsum_s[16];
    __shared__ int      sel_bin[16];
    __shared__ unsigned sel_cum[16];
    __shared__ float    out_s[16 * DD];

    const int tid  = threadIdx.x;
    const int w    = tid >> 5;                // wave id 0..15
    const int lane = tid & 31;
    const int b    = blockIdx.x >> 7;         // batch
    const int row0 = (blockIdx.x & 127) << 4; // 16-row tile base
    const int lh   = lane >> 4;               // lane half (0/1)
    const int ll   = lane & 15;

    // ---------------- Phase 1: S = Q * K^T (f16 WMMA, fp32 acc) -------------
    // A-frag (Q rows), per ISA 16-bit A 16x32 layout:
    // lane<16: M=lane, K=0..7 & 16..23 ; lane>=16: M=lane-16, K=8..15 & 24..31
    v16h aq0, aq1;
    {
        const float* qp = qg + (size_t)(b * NN + row0 + ll) * DD + lh * 8;
#pragma unroll
        for (int t = 0; t < 8; ++t) {
            aq0[t]     = (_Float16)qp[t];        // d = off+t
            aq0[8 + t] = (_Float16)qp[16 + t];   // d = off+16+t
            aq1[t]     = (_Float16)qp[32 + t];   // chunk 1
            aq1[8 + t] = (_Float16)qp[48 + t];
        }
    }
    // Each wave computes 8 contiguous 16x16 score tiles.
    for (int i = 0; i < 8; ++i) {
        const int n0 = (w * 8 + i) * 16;
        // B-frag (K rows as columns of K^T): lane holds col n=ll,
        // k(dim d) = lh*8+t (t<8), 16+lh*8+(t-8) (t>=8) within the 32-chunk.
        const float* kp = kg + (size_t)(b * NN + n0 + ll) * DD + lh * 8;
        v16h bk0, bk1;
#pragma unroll
        for (int t = 0; t < 16; ++t) {
            bk0[t] = (_Float16)kp[t + (t & 8)];
            bk1[t] = (_Float16)kp[32 + t + (t & 8)];
        }
        v8f cs = {};
        cs = __builtin_amdgcn_wmma_f32_16x16x32_f16(false, aq0, false, bk0,
                                                    (short)0, cs, false, false);
        cs = __builtin_amdgcn_wmma_f32_16x16x32_f16(false, aq1, false, bk1,
                                                    (short)0, cs, false, false);
        // C layout: lane<16 -> (M=j, N=lane) ; lane>=16 -> (M=8+j, N=lane-16)
#pragma unroll
        for (int j = 0; j < 8; ++j)
            sc[(lh * 8 + j) * SROW + n0 + ll] = cs[j];
    }
    for (int i = tid; i < 16 * DD; i += 512) out_s[i] = 0.0f;
    __syncthreads();

    // ------------- Phase 2: rank-1536 threshold via radix select ------------
    // One wave per row. Monotone key: u = f<0 ? ~bits : bits|0x80000000.
    const int knnv = knnp[0];
    unsigned prefix = 0;
    if (knnv) {
        int remaining = KEEP;
        unsigned* h = hist + w * 256;
        for (int shift = 24; shift >= 0; shift -= 8) {
            for (int i2 = lane; i2 < 256; i2 += 32) h[i2] = 0u;
            __syncthreads();
            const unsigned mhi = (shift == 24) ? 0u
                                               : (0xFFFFFFFFu << (shift + 8));
            for (int n = lane; n < NN; n += 32) {
                unsigned u = __float_as_uint(sc[w * SROW + n]);
                u = (u & 0x80000000u) ? ~u : (u | 0x80000000u);
                if ((u & mhi) == (prefix & mhi))
                    atomicAdd(&h[(u >> shift) & 255u], 1u);
            }
            __syncthreads();
            if (lane == 0) {
                unsigned cum = 0; int bsel = 0;
                for (int bb = 255; bb >= 0; --bb) {
                    unsigned c = h[bb];
                    if (cum + c >= (unsigned)remaining) { bsel = bb; break; }
                    cum += c;
                }
                sel_bin[w] = bsel; sel_cum[w] = cum;
            }
            __syncthreads();
            prefix |= ((unsigned)sel_bin[w]) << shift;
            remaining -= (int)sel_cum[w];
        }
    }
    if (lane == 0) thr_s[w] = knnv ? prefix : 0u;
    __syncthreads();

    // ------------- Phase 3a: masked softmax numerators (per row) ------------
    {
        const unsigned th = thr_s[w];
        float mx = -3.402823466e38f;
        for (int n = lane; n < NN; n += 32)
            mx = fmaxf(mx, sc[w * SROW + n]);   // row max is always kept
#pragma unroll
        for (int o = 16; o >= 1; o >>= 1) mx = fmaxf(mx, __shfl_xor(mx, o, 32));
        float sum = 0.0f;
        for (int n = lane; n < NN; n += 32) {
            const float s = sc[w * SROW + n];
            unsigned u = __float_as_uint(s);
            u = (u & 0x80000000u) ? ~u : (u | 0x80000000u);
            const float p = (u >= th) ? __expf(s - mx) : 0.0f;
            sc[w * SROW + n] = p;               // store unnormalized prob
            sum += p;
        }
#pragma unroll
        for (int o = 16; o >= 1; o >>= 1) sum += __shfl_xor(sum, o, 32);
        if (lane == 0) rowsum_s[w] = sum;
    }
    __syncthreads();

    // ------------- Phase 3b: O = P * V via WMMA, waves split n --------------
    v8f acc0 = {}, acc1 = {}, acc2 = {}, acc3 = {};
    for (int cc = 0; cc < 4; ++cc) {
        const int nb = w * 128 + cc * 32;       // this wave's n-chunk
        // A-frag from P strip (row = ll, K = n offsets nb+lh*16+t, contiguous)
        v16h ap;
        {
            const int off = nb + lh * 16;
#pragma unroll
            for (int t = 0; t < 8; ++t) {
                ap[t]     = (_Float16)sc[ll * SROW + off + t];
                ap[8 + t] = (_Float16)sc[ll * SROW + off + 8 + t];
            }
        }
        // B-frags from V: lane holds column dcol=g*16+ll, K rows nb+lh*16..+15
        const float* vp = vg + (size_t)(b * NN + nb + lh * 16) * DD + ll;
#pragma unroll
        for (int g = 0; g < 4; ++g) {
            v16h bv;
            const float* vpg = vp + g * 16;
#pragma unroll
            for (int t = 0; t < 16; ++t) bv[t] = (_Float16)vpg[(size_t)t * DD];
            v8f& acc = (g == 0) ? acc0 : (g == 1) ? acc1 : (g == 2) ? acc2 : acc3;
            acc = __builtin_amdgcn_wmma_f32_16x16x32_f16(false, ap, false, bv,
                                                         (short)0, acc, false, false);
        }
    }
    // Reduce wave partials into the 16x64 tile via LDS float atomics.
#pragma unroll
    for (int g = 0; g < 4; ++g) {
        const v8f& acc = (g == 0) ? acc0 : (g == 1) ? acc1 : (g == 2) ? acc2 : acc3;
        const int dcol = g * 16 + ll;
#pragma unroll
        for (int j = 0; j < 8; ++j)
            atomicAdd(&out_s[(lh * 8 + j) * DD + dcol], acc[j]);
    }
    __syncthreads();

    // ---------------- normalize and store -----------------------------------
    for (int i = tid; i < 16 * DD; i += 512) {
        const int m = i >> 6, d = i & 63;
        outg[(size_t)(b * NN + row0 + m) * DD + d] = out_s[i] / rowsum_s[m];
    }
}

extern "C" void kernel_launch(void* const* d_in, const int* in_sizes, int n_in,
                              void* d_out, int out_size, void* d_ws, size_t ws_size,
                              hipStream_t stream) {
    (void)in_sizes; (void)n_in; (void)out_size; (void)d_ws; (void)ws_size;
    const float* q   = (const float*)d_in[0];
    const float* k   = (const float*)d_in[1];
    const float* v   = (const float*)d_in[2];
    const int*   knn = (const int*)d_in[3];
    float* out = (float*)d_out;
    dim3 grid(NB * (NN / 16));   // 2048 workgroups: (batch, 16-row tile)
    dim3 block(512);             // 16 wave32 waves
    attn_topk_kernel<<<grid, block, 0, stream>>>(q, k, v, knn, out);
}
